// GATLayer_9603546874158
// MI455X (gfx1250) — matrix operand order repaired and checked by reading, not compiled, not run on hardware
//
#include <hip/hip_runtime.h>

// ---------------------------------------------------------------------------
// GAT layer forward for MI455X (gfx1250, wave32, WMMA).
// h = x @ W^T ; q = h @ A_h ; e = q @ h^T ; softmax(leaky(e)) ; out = attn @ h
// Big einsums fused flash-attention style; bf16 WMMA, fp32 accumulate.
// Round 3: sched_group_barrier-pinned software pipelines (the backend RA was
// collapsing source-level double buffering into full s_wait_loadcnt 0 drains).
// ---------------------------------------------------------------------------

typedef __bf16 bf16_t;
typedef __attribute__((ext_vector_type(16))) __bf16 bf16x16;
typedef __attribute__((ext_vector_type(8)))  float  fx8;

#define NN    4096
#define FIN   256
#define FOUT  512
#define HEADS 8
#define DH    64
#define NEG_SLOPE 0.2f

// sched_group_barrier masks
#define SGB_MFMA  0x008   // MFMA/WMMA pipe
#define SGB_VMRD  0x020   // VMEM reads

// float -> bf16 (native convert; RNE)
__device__ __forceinline__ bf16_t f2bf(float f) { return (bf16_t)f; }

// load 8 contiguous bf16 (16 bytes) into half of a fragment
__device__ __forceinline__ void ld16B(bf16x16& dst, int half, const bf16_t* p) {
  uint4 t = *(const uint4*)p;
  __builtin_memcpy(((char*)&dst) + half * 16, &t, 16);
}

// A fragment 16x32 (MxK) from row-major source (ld elems).
// ISA layout: lane l(0-15),hi: row M=l, K-runs [k0+8hi,+8) and [k0+16+8hi,+8).
__device__ __forceinline__ bf16x16 load_afrag(const bf16_t* base, int ld,
                                              int row0, int k0, int l, int hi) {
  bf16x16 f;
  const bf16_t* r = base + (size_t)(row0 + l) * ld + k0 + hi * 8;
  ld16B(f, 0, r);
  ld16B(f, 1, r + 16);
  return f;
}

// B fragment 32x16 (KxN) from B^T stored row-major (rows = N, contiguous K).
// ISA layout: lane l,hi: column N=l, K-runs [k0+8hi,+8) and [k0+16+8hi,+8).
__device__ __forceinline__ bf16x16 load_bfrag(const bf16_t* baseT, int ld,
                                              int n0, int k0, int l, int hi) {
  bf16x16 f;
  const bf16_t* r = baseT + (size_t)(n0 + l) * ld + k0 + hi * 8;
  ld16B(f, 0, r);
  ld16B(f, 1, r + 16);
  return f;
}

__device__ __forceinline__ fx8 wmma_bf16(bf16x16 a, bf16x16 b, fx8 c) {
  // (neg_a, A, neg_b, B, c_mod, C, reuse_a, reuse_b)
  return __builtin_amdgcn_wmma_f32_16x16x32_bf16(false, a, false, b,
                                                 (short)0, c, false, false);
}

// ---------------------------------------------------------------------------
// Kernel 0: fp32 -> bf16 conversions. AbT[h][e][d] = A[h][d][e] (B^T layout).
// ---------------------------------------------------------------------------
__global__ __launch_bounds__(256) void cvt_kernel(const float* __restrict__ x,
                                                  const float* __restrict__ W,
                                                  const float* __restrict__ A,
                                                  bf16_t* xb, bf16_t* Wb, bf16_t* AbT) {
  const int t1 = NN * FIN, t2 = FOUT * FIN, t3 = HEADS * DH * DH;
  const int total = t1 + t2 + t3;
  for (int i = blockIdx.x * blockDim.x + threadIdx.x; i < total;
       i += gridDim.x * blockDim.x) {
    if (i < t1) {
      xb[i] = f2bf(x[i]);
    } else if (i < t1 + t2) {
      int j = i - t1;
      Wb[j] = f2bf(W[j]);
    } else {
      int j = i - t1 - t2;
      int hh = j >> 12, e = (j >> 6) & 63, dd = j & 63;
      AbT[j] = f2bf(A[hh * 4096 + dd * 64 + e]);
    }
  }
}

// ---------------------------------------------------------------------------
// Kernel 1: h = x @ W^T. Each wave: 16 rows x 64 cols, K=256 (8 WMMA k-steps).
// sched_group_barrier pins a depth-1 software pipeline:
//   VMEM x20, (WMMA x4, VMEM x10) x6, WMMA x8
// so k-step ks+1 loads are in flight across k-step ks WMMAs (partial waits).
// Writes hb[head][n][d] and transposed hbT[head][d][n] (both bf16).
// ---------------------------------------------------------------------------
__global__ __launch_bounds__(128) void linear_kernel(const bf16_t* __restrict__ xb,
                                                     const bf16_t* __restrict__ Wb,
                                                     bf16_t* __restrict__ hb,
                                                     bf16_t* __restrict__ hbT) {
  const int wave = threadIdx.x >> 5, lane = threadIdx.x & 31;
  const int l = lane & 15, hi = lane >> 4;
  const int n0 = (blockIdx.x * 4 + wave) * 16;
  const int c0 = blockIdx.y * 64;

  const fx8 zero = {0.f, 0.f, 0.f, 0.f, 0.f, 0.f, 0.f, 0.f};
  fx8 acc[4];
  acc[0] = acc[1] = acc[2] = acc[3] = zero;

  // prologue: fragments for k-step 0
  bf16x16 a_cur = load_afrag(xb, FIN, n0, 0, l, hi);
  bf16x16 b_cur[4];
#pragma unroll
  for (int cs = 0; cs < 4; ++cs)
    b_cur[cs] = load_bfrag(Wb, FIN, c0 + cs * 16, 0, l, hi);

#pragma unroll
  for (int ks = 0; ks < FIN / 32; ++ks) {
    bf16x16 a_nxt = a_cur;
    bf16x16 b_nxt[4];
    if (ks < FIN / 32 - 1) {
      a_nxt = load_afrag(xb, FIN, n0, (ks + 1) * 32, l, hi);
#pragma unroll
      for (int cs = 0; cs < 4; ++cs)
        b_nxt[cs] = load_bfrag(Wb, FIN, c0 + cs * 16, (ks + 1) * 32, l, hi);
    } else {
#pragma unroll
      for (int cs = 0; cs < 4; ++cs) b_nxt[cs] = b_cur[cs];
    }
#pragma unroll
    for (int cs = 0; cs < 4; ++cs)
      acc[cs] = wmma_bf16(a_cur, b_cur[cs], acc[cs]);
    a_cur = a_nxt;
#pragma unroll
    for (int cs = 0; cs < 4; ++cs) b_cur[cs] = b_nxt[cs];
  }

  // Pin the pipeline: 80 VMEM reads total, 32 WMMAs.
  // loads(ks0..1) ; then wmma(ks) overlapped with loads(ks+2).
  __builtin_amdgcn_sched_group_barrier(SGB_VMRD, 20, 0);
#pragma unroll
  for (int i = 0; i < 6; ++i) {
    __builtin_amdgcn_sched_group_barrier(SGB_MFMA, 4, 0);
    __builtin_amdgcn_sched_group_barrier(SGB_VMRD, 10, 0);
  }
  __builtin_amdgcn_sched_group_barrier(SGB_MFMA, 8, 0);

#pragma unroll
  for (int cs = 0; cs < 4; ++cs) {
#pragma unroll
    for (int r = 0; r < 8; ++r) {
      int c = c0 + cs * 16 + l;
      int head = c >> 6, dd = c & 63;
      int n = n0 + r + 8 * hi;
      bf16_t v = f2bf(acc[cs][r]);
      hb [((size_t)head * NN + n) * DH + dd] = v;
      hbT[((size_t)head * DH + dd) * NN + n] = v;
    }
  }
}

// ---------------------------------------------------------------------------
// Kernel 2: q[n][h][e] = sum_d h[n][h][d] * A[h][d][e]. Wave: 16 rows x 64 cols.
// ---------------------------------------------------------------------------
__global__ __launch_bounds__(128) void qproj_kernel(const bf16_t* __restrict__ hb,
                                                    const bf16_t* __restrict__ AbT,
                                                    bf16_t* __restrict__ qb) {
  const int wave = threadIdx.x >> 5, lane = threadIdx.x & 31;
  const int l = lane & 15, hi = lane >> 4;
  const int head = blockIdx.y;
  const int n0 = (blockIdx.x * 4 + wave) * 16;

  const bf16_t* h_h = hb + (size_t)head * NN * DH;
  const bf16_t* aT = AbT + (size_t)head * DH * DH;

  const fx8 zero = {0.f, 0.f, 0.f, 0.f, 0.f, 0.f, 0.f, 0.f};
  fx8 acc[4];
  acc[0] = acc[1] = acc[2] = acc[3] = zero;

  // issue all fragment loads up front (K = 64 -> 2 k-steps), then compute
  bf16x16 a0 = load_afrag(h_h, DH, n0, 0, l, hi);
  bf16x16 a1 = load_afrag(h_h, DH, n0, 32, l, hi);
  bf16x16 b0[4], b1[4];
#pragma unroll
  for (int cs = 0; cs < 4; ++cs) {
    b0[cs] = load_bfrag(aT, DH, cs * 16, 0, l, hi);
    b1[cs] = load_bfrag(aT, DH, cs * 16, 32, l, hi);
  }
#pragma unroll
  for (int cs = 0; cs < 4; ++cs) {
    acc[cs] = wmma_bf16(a0, b0[cs], acc[cs]);
    acc[cs] = wmma_bf16(a1, b1[cs], acc[cs]);
  }
  __builtin_amdgcn_sched_group_barrier(SGB_VMRD, 20, 0);
  __builtin_amdgcn_sched_group_barrier(SGB_MFMA, 8, 0);

#pragma unroll
  for (int cs = 0; cs < 4; ++cs) {
#pragma unroll
    for (int r = 0; r < 8; ++r) {
      int n = n0 + r + 8 * hi;
      qb[((size_t)head * NN + n) * DH + cs * 16 + l] = f2bf(acc[cs][r]);
    }
  }
}

// ---------------------------------------------------------------------------
// Kernel 3: fused scores + LeakyReLU + online softmax + P@V (flash style).
// Grid (NN/64, HEADS), 4 waves per WG, each wave owns 16 query rows.
// Per 64-key block: 8 score WMMAs + 8 P@V WMMAs. Score tensor never hits HBM.
// sched_group_barrier pins ALL 32 fragment loads (K and V) ahead of the score
// WMMAs, so V-load latency is hidden under the whole softmax VALU/TRANS work.
// ---------------------------------------------------------------------------
__global__ __launch_bounds__(128) void flash_kernel(const bf16_t* __restrict__ qb,
                                                    const bf16_t* __restrict__ hb,
                                                    const bf16_t* __restrict__ hbT,
                                                    float* __restrict__ out) {
  const int wave = threadIdx.x >> 5, lane = threadIdx.x & 31;
  const int l = lane & 15, hi = lane >> 4;
  const int head = blockIdx.y;
  const int n0 = blockIdx.x * 64 + wave * 16;

  const bf16_t* q_h  = qb  + (size_t)head * NN * DH;
  const bf16_t* k_h  = hb  + (size_t)head * NN * DH;
  const bf16_t* vT_h = hbT + (size_t)head * DH * NN;

  // Q fragments stay resident for the whole key loop (K = 64 -> 2 k-steps)
  bf16x16 qf0 = load_afrag(q_h, DH, n0, 0, l, hi);
  bf16x16 qf1 = load_afrag(q_h, DH, n0, 32, l, hi);

  const fx8 zero = {0.f, 0.f, 0.f, 0.f, 0.f, 0.f, 0.f, 0.f};
  fx8 acc[4];
  acc[0] = acc[1] = acc[2] = acc[3] = zero;
  float mrow[8], lrow[8];
#pragma unroll
  for (int r = 0; r < 8; ++r) { mrow[r] = -__builtin_inff(); lrow[r] = 0.f; }

  // per-wave 16x64 P tile: C/D layout -> A layout bounce through LDS
  __shared__ __align__(16) bf16_t pbuf[4][16 * 64];
  bf16_t* pw = pbuf[wave];

  for (int m0 = 0; m0 < NN; m0 += 64) {
    // ---- V fragments first: nothing depends on them until P@V, so the
    //      whole softmax pipeline below runs while they are in flight ----
    bf16x16 vf0[4], vf1[4];
#pragma unroll
    for (int ds = 0; ds < 4; ++ds) {
      vf0[ds] = load_bfrag(vT_h, NN, ds * 16, m0, l, hi);
      vf1[ds] = load_bfrag(vT_h, NN, ds * 16, m0 + 32, l, hi);
    }
    // ---- scores s[n][m] = q . k ----
    fx8 s[4];
#pragma unroll
    for (int cs = 0; cs < 4; ++cs) {
      bf16x16 b0 = load_bfrag(k_h, DH, m0 + cs * 16, 0, l, hi);
      bf16x16 b1 = load_bfrag(k_h, DH, m0 + cs * 16, 32, l, hi);
      fx8 c = zero;
      c = wmma_bf16(qf0, b0, c);
      c = wmma_bf16(qf1, b1, c);
      s[cs] = c;
    }
    // ---- LeakyReLU(0.2) ----
#pragma unroll
    for (int cs = 0; cs < 4; ++cs) {
#pragma unroll
      for (int r = 0; r < 8; ++r) {
        float v = s[cs][r];
        s[cs][r] = v >= 0.f ? v : NEG_SLOPE * v;
      }
    }
    // ---- online softmax: new row max (row spans 16 lanes of one half) ----
    float newm[8], alpha[8], rs[8];
#pragma unroll
    for (int r = 0; r < 8; ++r) {
      float mx = fmaxf(fmaxf(s[0][r], s[1][r]), fmaxf(s[2][r], s[3][r]));
#pragma unroll
      for (int m = 1; m < 16; m <<= 1) mx = fmaxf(mx, __shfl_xor(mx, m, 32));
      newm[r] = fmaxf(mrow[r], mx);
      alpha[r] = __expf(mrow[r] - newm[r]);
      rs[r] = 0.f;
    }
    // ---- p = exp(s - newm): row sums + spill to LDS in [row][m] layout ----
#pragma unroll
    for (int cs = 0; cs < 4; ++cs) {
#pragma unroll
      for (int r = 0; r < 8; ++r) {
        float p = __expf(s[cs][r] - newm[r]);
        rs[r] += p;
        pw[(r + 8 * hi) * 64 + cs * 16 + l] = f2bf(p);
      }
    }
#pragma unroll
    for (int r = 0; r < 8; ++r) {
      float t = rs[r];
#pragma unroll
      for (int m = 1; m < 16; m <<= 1) t += __shfl_xor(t, m, 32);
      lrow[r] = lrow[r] * alpha[r] + t;
      mrow[r] = newm[r];
    }
    // ---- rescale running output ----
#pragma unroll
    for (int ds = 0; ds < 4; ++ds) {
#pragma unroll
      for (int r = 0; r < 8; ++r) acc[ds][r] *= alpha[r];
    }
    // ---- O += P @ V (DS ops from the same wave are in-order) ----
    bf16x16 pf0 = load_afrag(pw, 64, 0, 0, l, hi);
    bf16x16 pf1 = load_afrag(pw, 64, 0, 32, l, hi);
#pragma unroll
    for (int ds = 0; ds < 4; ++ds) {
      acc[ds] = wmma_bf16(pf0, vf0[ds], acc[ds]);
      acc[ds] = wmma_bf16(pf1, vf1[ds], acc[ds]);
    }
    // Pin: all 32 b128 fragment loads (V then K) issue before the score
    // WMMAs; the two WMMA clusters stay in order. Softmax VALU/TRANS/DS
    // are left unconstrained (their data deps already order them) and
    // execute while nothing else is pending.
    __builtin_amdgcn_sched_group_barrier(SGB_VMRD, 32, 0);
    __builtin_amdgcn_sched_group_barrier(SGB_MFMA, 8, 0);
    __builtin_amdgcn_sched_group_barrier(SGB_MFMA, 8, 0);
  }
  // ---- epilogue: divide by softmax denominator, write fp32 output ----
#pragma unroll
  for (int ds = 0; ds < 4; ++ds) {
#pragma unroll
    for (int r = 0; r < 8; ++r) {
      int n = n0 + r + 8 * hi;
      out[(size_t)n * FOUT + head * DH + ds * 16 + l] = acc[ds][r] / lrow[r];
    }
  }
}

// ---------------------------------------------------------------------------
// Launcher. Inputs: x[4096,256] f32, adj (unused), W[512,256] f32, A[8,64,64] f32.
// Workspace (~14.3 MB): bf16 copies of x/W/A^T, h, h^T, q.
// ---------------------------------------------------------------------------
extern "C" void kernel_launch(void* const* d_in, const int* in_sizes, int n_in,
                              void* d_out, int out_size, void* d_ws, size_t ws_size,
                              hipStream_t stream) {
  const float* x = (const float*)d_in[0];
  // d_in[1] = adj, unused by the reference forward
  const float* W = (const float*)d_in[2];
  const float* A = (const float*)d_in[3];
  float* out = (float*)d_out;

  char* p = (char*)d_ws;
  bf16_t* xb  = (bf16_t*)p; p += (size_t)NN * FIN * 2;
  bf16_t* Wb  = (bf16_t*)p; p += (size_t)FOUT * FIN * 2;
  bf16_t* AbT = (bf16_t*)p; p += (size_t)HEADS * DH * DH * 2;
  bf16_t* hb  = (bf16_t*)p; p += (size_t)HEADS * NN * DH * 2;
  bf16_t* hbT = (bf16_t*)p; p += (size_t)HEADS * DH * NN * 2;
  bf16_t* qb  = (bf16_t*)p; p += (size_t)HEADS * NN * DH * 2;

  cvt_kernel<<<1024, 256, 0, stream>>>(x, W, A, xb, Wb, AbT);
  linear_kernel<<<dim3(64, 8), 128, 0, stream>>>(xb, Wb, hb, hbT);
  qproj_kernel<<<dim3(64, 8), 128, 0, stream>>>(hb, AbT, qb);
  flash_kernel<<<dim3(64, 8), 128, 0, stream>>>(qb, hb, hbT, out);
}
